// HSPA_18691697672777
// MI455X (gfx1250) — compile-verified, matrix operand order checked by reading, not run on hardware
//
#include <hip/hip_runtime.h>
#include <hip/hip_bf16.h>

// ---------------------------------------------------------------------------
// HSPA sparse-attention block for MI455X (gfx1250, wave32, WMMA).
//
//   e1 = PReLU(x W1 + b1)           [B,N,128]  (queries)
//   e2 = PReLU(x W2 + b2)           [B,N,128]  (keys)
//   v  = PReLU(x Wa + ba)           [B,N,256]  (values, stored transposed)
//   S  = e1 e2^T                    [16 x 4096] stripe, kept in 256KB LDS
//   P  = sparsemax(S, dim=-1)       tau via 24-step bisection (no sort)
//   out= P v + x
//
// All GEMMs use v_wmma_f32_16x16x32_bf16 (f32 accumulate). Per-batch K/V
// working set (~3MB bf16) is L2-resident; the N x N score matrix never
// touches HBM. Projection A-operands staged in LDS via the gfx1250 async
// global->LDS path (ASYNCcnt); attention GEMMs use register double
// buffering so WMMA issue overlaps the next tile's global loads.
// ---------------------------------------------------------------------------

#define B_   4
#define N_   4096
#define C_   256
#define Cr_  128

typedef __attribute__((ext_vector_type(16))) __bf16 v16bf;
typedef __attribute__((ext_vector_type(8)))  __bf16 v8bf;
typedef __attribute__((ext_vector_type(8)))  float  v8f;

__device__ __forceinline__ v8f wmma_bf16(v16bf a, v16bf b, v8f c) {
    // 8 args: (neg_a, A, neg_b, B, c_mod, C, reuse_a, reuse_b)
    return __builtin_amdgcn_wmma_f32_16x16x32_bf16(false, a, false, b,
                                                   (short)0, c, false, false);
}

// A/B fragment loader for 16x16x32 bf16 WMMA.
// A (16x32): lane L<16 -> row L, K {0..7, 16..23}; lanes 16..31 -> K {8..15, 24..31}.
// B (32x16): same pattern with lane = output column. Both reduce to two
// contiguous 16-byte loads from a row-major bf16 matrix with leading dim `ld`.
__device__ __forceinline__ v16bf load_frag(const __bf16* p, int ld, int rc0,
                                           int kc, int lane) {
    const int rc = rc0 + (lane & 15);
    const int k  = kc + ((lane & 16) ? 8 : 0);
    const __bf16* base = p + (size_t)rc * ld + k;
    v8bf lo = *(const v8bf*)(base);
    v8bf hi = *(const v8bf*)(base + 16);
    v16bf r;
#pragma unroll
    for (int j = 0; j < 8; ++j) { r[j] = lo[j]; r[j + 8] = hi[j]; }
    return r;
}

// gfx1250 async global->LDS copy (ASYNCcnt-tracked). lds_off is the byte
// offset inside the LDS segment; gaddr is a per-lane global address.
__device__ __forceinline__ void async_copy_b128(unsigned lds_off,
                                                const void* gaddr) {
    asm volatile("global_load_async_to_lds_b128 %0, %1, off"
                 :: "v"(lds_off), "v"(gaddr) : "memory");
}
__device__ __forceinline__ void wait_asynccnt0() {
    asm volatile("s_wait_asynccnt 0" ::: "memory");
}
__device__ __forceinline__ unsigned lds_off_of(void* p) {
    return (unsigned)(uintptr_t)(__attribute__((address_space(3))) char*)p;
}

// ---------------------------------------------------------------------------
// Kernel 0: convert x -> bf16; transpose+convert W1,W2,Wa -> bf16 [out_dim, C]
// so B-fragments of the projection GEMMs are contiguous along C.
// ---------------------------------------------------------------------------
__global__ void hspa_prep_kernel(const float* __restrict__ x,
                                 const float* __restrict__ W1,
                                 const float* __restrict__ W2,
                                 const float* __restrict__ Wa,
                                 __bf16* __restrict__ xbf,
                                 __bf16* __restrict__ w1t,
                                 __bf16* __restrict__ w2t,
                                 __bf16* __restrict__ wat) {
    size_t i = (size_t)blockIdx.x * blockDim.x + threadIdx.x;
    const size_t NX = (size_t)B_ * N_ * C_;
    if (i < NX) { xbf[i] = (__bf16)x[i]; return; }
    i -= NX;
    if (i < (size_t)C_ * Cr_) {
        int c = (int)(i / Cr_), d = (int)(i % Cr_);
        w1t[(size_t)d * C_ + c] = (__bf16)W1[i];
        return;
    }
    i -= (size_t)C_ * Cr_;
    if (i < (size_t)C_ * Cr_) {
        int c = (int)(i / Cr_), d = (int)(i % Cr_);
        w2t[(size_t)d * C_ + c] = (__bf16)W2[i];
        return;
    }
    i -= (size_t)C_ * Cr_;
    if (i < (size_t)C_ * C_) {
        int c = (int)(i / C_), d = (int)(i % C_);
        wat[(size_t)d * C_ + c] = (__bf16)Wa[i];
    }
}

// ---------------------------------------------------------------------------
// Kernel 1: the three 1x1-conv projections, fully on WMMA.
// Block = 256 threads (8 waves) owns 64 rows x all 512 output columns.
// The 64x256 bf16 x-tile (32KB) is staged ONCE into LDS with async
// global->LDS copies; all A-fragments then come from ds_load_b128, cutting
// A-operand L2 traffic 32x (each row is consumed by 32 column tiles).
// v is stored transposed [C, N] so the attention B-fragments stream
// contiguously along keys.
// ---------------------------------------------------------------------------
__global__ __launch_bounds__(256) void hspa_proj_kernel(
        const __bf16* __restrict__ xbf,
        const __bf16* __restrict__ w1t,
        const __bf16* __restrict__ w2t,
        const __bf16* __restrict__ wat,
        const float* __restrict__ b1, const float* __restrict__ a1,
        const float* __restrict__ b2, const float* __restrict__ a2,
        const float* __restrict__ ba, const float* __restrict__ aa,
        __bf16* __restrict__ e1, __bf16* __restrict__ e2,
        __bf16* __restrict__ vt) {
    __shared__ __bf16 xs[64 * C_];          // 32 KB x-tile

    const int b    = blockIdx.y;
    const int m0   = blockIdx.x * 64;
    const int tid  = threadIdx.x;
    const int wave = tid >> 5;
    const int lane = tid & 31;

    // ---- async stage x-tile into LDS (2048 16B chunks / 256 threads) -----
    {
        const __bf16* src = xbf + ((size_t)b * N_ + m0) * C_;
        const unsigned base = lds_off_of(xs);
#pragma unroll
        for (int i = 0; i < 8; ++i) {
            const int elt = (tid + i * 256) * 8;        // 8 bf16 per chunk
            async_copy_b128(base + elt * 2, src + elt);
        }
        wait_asynccnt0();
    }
    __syncthreads();

#pragma unroll
    for (int i = 0; i < 4; ++i) {
        const int t = wave * 4 + i;
        const __bf16* wt;
        const float* bias;
        float alpha;
        int kind, ct;
        if (t < 8)       { wt = w1t; bias = b1; alpha = a1[0]; kind = 0; ct = t; }
        else if (t < 16) { wt = w2t; bias = b2; alpha = a2[0]; kind = 1; ct = t - 8; }
        else             { wt = wat; bias = ba; alpha = aa[0]; kind = 2; ct = t - 16; }
        const int n0   = ct * 16;
        const int col  = n0 + (lane & 15);
        const float bc = bias[col];

        for (int rt = 0; rt < 4; ++rt) {
            const int rowbase = rt * 16;               // local row in xs
            v8f acc = {};
#pragma unroll
            for (int kc = 0; kc < 8; ++kc) {
                v16bf a  = load_frag(xs, C_, rowbase, kc * 32, lane);
                v16bf bm = load_frag(wt, C_, n0,      kc * 32, lane);
                acc = wmma_bf16(a, bm, acc);
            }
            const int mh = (lane & 16) ? 8 : 0;
            if (kind < 2) {
                __bf16* eo = (kind == 0 ? e1 : e2) + (size_t)b * N_ * Cr_;
#pragma unroll
                for (int j = 0; j < 8; ++j) {
                    float v = acc[j] + bc;
                    v = (v >= 0.f) ? v : alpha * v;
                    eo[(size_t)(m0 + rowbase + mh + j) * Cr_ + col] = (__bf16)v;
                }
            } else {
                v8bf pk;
#pragma unroll
                for (int j = 0; j < 8; ++j) {
                    float v = acc[j] + bc;
                    v = (v >= 0.f) ? v : alpha * v;
                    pk[j] = (__bf16)v;
                }
                *(v8bf*)(vt + ((size_t)b * C_ + col) * N_ + m0 + rowbase + mh) = pk;
            }
        }
    }
}

// ---------------------------------------------------------------------------
// Kernel 2: fused attention. One block = 16 query rows.
//   Phase A: S[16,4096] = Q K^T via WMMA into 256KB dynamic LDS (8 waves
//            partition the 4096 keys, 512 each). K-fragments register
//            double-buffered: tile jt+1's loads issue before tile jt's WMMAs.
//   Phase B: sparsemax tau per row by bisection on f(t)=sum(relu(s-t))-1,
//            t in [rowmax-1, rowmax]; 16 lanes/row, shfl_xor reductions.
//   Phase C: out = relu(S - tau) V^T via WMMA (A rebuilt from LDS with the
//            threshold applied at fragment load; V-fragments double-buffered),
//            + residual x.
// ---------------------------------------------------------------------------
__global__ __launch_bounds__(256) void hspa_attn_kernel(
        const __bf16* __restrict__ e1,
        const __bf16* __restrict__ e2,
        const __bf16* __restrict__ vt,
        const float* __restrict__ x,
        float* __restrict__ out) {
    extern __shared__ float smem[];
    float* S    = smem;             // 16 * 4096 f32 scores
    float* taub = smem + 16 * N_;   // 16 taus

    const int b       = blockIdx.y;
    const int rowbase = blockIdx.x * 16;
    const int tid     = threadIdx.x;
    const int wave    = tid >> 5;
    const int lane    = tid & 31;

    const __bf16* e1b = e1 + (size_t)b * N_ * Cr_;
    const __bf16* e2b = e2 + (size_t)b * N_ * Cr_;
    const __bf16* vtb = vt + (size_t)b * C_ * N_;

    // ---- Phase A: scores (double-buffered K fragments) -------------------
    v16bf qf[4];
#pragma unroll
    for (int kc = 0; kc < 4; ++kc)
        qf[kc] = load_frag(e1b, Cr_, rowbase, kc * 32, lane);

    v16bf kf[2][4];
#pragma unroll
    for (int kc = 0; kc < 4; ++kc)
        kf[0][kc] = load_frag(e2b, Cr_, wave * 512, kc * 32, lane);

    for (int jt = 0; jt < 32; ++jt) {
        const int cur = jt & 1, nxt = cur ^ 1;
        if (jt + 1 < 32) {
            const int n1 = wave * 512 + (jt + 1) * 16;
            __builtin_prefetch(e2b + (size_t)(n1 + 16) * Cr_, 0, 1);
#pragma unroll
            for (int kc = 0; kc < 4; ++kc)
                kf[nxt][kc] = load_frag(e2b, Cr_, n1, kc * 32, lane);
        }
        v8f acc = {};
#pragma unroll
        for (int kc = 0; kc < 4; ++kc)
            acc = wmma_bf16(qf[kc], kf[cur][kc], acc);

        const int n0   = wave * 512 + jt * 16;
        const int mh   = (lane & 16) ? 8 : 0;
        const int ncol = n0 + (lane & 15);
#pragma unroll
        for (int j = 0; j < 8; ++j)
            S[(size_t)(mh + j) * N_ + ncol] = acc[j];
    }
    __syncthreads();

    // ---- Phase B: sparsemax tau (bisection, no sort) ---------------------
    {
        const int row = tid >> 4;     // 16 threads per row
        const int sub = tid & 15;
        const float* Sr = S + (size_t)row * N_;

        float mx = -3.4e38f;
#pragma unroll 4
        for (int k = 0; k < 64; ++k) {
            const float4 v = *(const float4*)(Sr + k * 64 + sub * 4);
            mx = fmaxf(fmaxf(mx, v.x), fmaxf(fmaxf(v.y, v.z), v.w));
        }
#pragma unroll
        for (int off = 8; off; off >>= 1)
            mx = fmaxf(mx, __shfl_xor(mx, off, 16));

        float lo = mx - 1.0f, hi = mx;   // f(lo) >= 1, f(hi) = 0
        for (int it = 0; it < 24; ++it) {
            const float mid = 0.5f * (lo + hi);
            float s = 0.f;
#pragma unroll 4
            for (int k = 0; k < 64; ++k) {
                const float4 v = *(const float4*)(Sr + k * 64 + sub * 4);
                s += fmaxf(v.x - mid, 0.f) + fmaxf(v.y - mid, 0.f) +
                     fmaxf(v.z - mid, 0.f) + fmaxf(v.w - mid, 0.f);
            }
#pragma unroll
            for (int off = 8; off; off >>= 1)
                s += __shfl_xor(s, off, 16);
            if (s > 1.0f) lo = mid; else hi = mid;
        }
        if (sub == 0) taub[row] = 0.5f * (lo + hi);
    }
    __syncthreads();

    // ---- Phase C: out = P V^T + x (double-buffered V fragments) ----------
    const float taur = taub[lane & 15];   // A-fragment row for this lane
    const int ch0 = wave * 32, ch1 = ch0 + 16;
    const int arow = lane & 15;
    const int khh  = (lane & 16) ? 8 : 0;
    const float* sp0 = S + (size_t)arow * N_;

    v16bf vb[2][2];
    vb[0][0] = load_frag(vtb, N_, ch0, 0, lane);
    vb[0][1] = load_frag(vtb, N_, ch1, 0, lane);

    v8f acc0 = {}, acc1 = {};
    for (int kt = 0; kt < 128; ++kt) {
        const int cur = kt & 1, nxt = cur ^ 1;
        if (kt + 1 < 128) {
            vb[nxt][0] = load_frag(vtb, N_, ch0, (kt + 1) * 32, lane);
            vb[nxt][1] = load_frag(vtb, N_, ch1, (kt + 1) * 32, lane);
        }
        const float* sp = sp0 + kt * 32 + khh;
        v16bf af;
#pragma unroll
        for (int j = 0; j < 8; ++j) {
            af[j]     = (__bf16)fmaxf(sp[j]      - taur, 0.f);
            af[j + 8] = (__bf16)fmaxf(sp[j + 16] - taur, 0.f);
        }
        acc0 = wmma_bf16(af, vb[cur][0], acc0);
        acc1 = wmma_bf16(af, vb[cur][1], acc1);
    }

    const int mh = (lane & 16) ? 8 : 0;
    const int c0 = ch0 + (lane & 15);
    const float* xb = x   + ((size_t)b * N_ + rowbase) * C_;
    float*       ob = out + ((size_t)b * N_ + rowbase) * C_;
#pragma unroll
    for (int j = 0; j < 8; ++j) {
        const size_t o = (size_t)(mh + j) * C_ + c0;
        ob[o]      = acc0[j] + xb[o];
        ob[o + 16] = acc1[j] + xb[o + 16];
    }
}

// ---------------------------------------------------------------------------
extern "C" void kernel_launch(void* const* d_in, const int* in_sizes, int n_in,
                              void* d_out, int out_size, void* d_ws, size_t ws_size,
                              hipStream_t stream) {
    const float* x  = (const float*)d_in[0];
    const float* W1 = (const float*)d_in[1];
    const float* b1 = (const float*)d_in[2];
    const float* a1 = (const float*)d_in[3];
    const float* W2 = (const float*)d_in[4];
    const float* b2 = (const float*)d_in[5];
    const float* a2 = (const float*)d_in[6];
    const float* Wa = (const float*)d_in[7];
    const float* ba = (const float*)d_in[8];
    const float* aa = (const float*)d_in[9];
    float* out = (float*)d_out;

    // Workspace layout (bf16 staging, ~24.3 MB total)
    char* p = (char*)d_ws;
    __bf16* xbf = (__bf16*)p; p += (size_t)B_ * N_ * C_  * 2;   // 8 MB
    __bf16* w1t = (__bf16*)p; p += (size_t)Cr_ * C_      * 2;   // 64 KB
    __bf16* w2t = (__bf16*)p; p += (size_t)Cr_ * C_      * 2;   // 64 KB
    __bf16* wat = (__bf16*)p; p += (size_t)C_  * C_      * 2;   // 128 KB
    __bf16* e1  = (__bf16*)p; p += (size_t)B_ * N_ * Cr_ * 2;   // 4 MB
    __bf16* e2  = (__bf16*)p; p += (size_t)B_ * N_ * Cr_ * 2;   // 4 MB
    __bf16* vt  = (__bf16*)p; p += (size_t)B_ * C_ * N_  * 2;   // 8 MB

    // Kernel 0: bf16 conversion + weight transposes
    {
        const size_t total = (size_t)B_ * N_ * C_ + 2u * C_ * Cr_ + (size_t)C_ * C_;
        const int blocks = (int)((total + 255) / 256);
        hspa_prep_kernel<<<blocks, 256, 0, stream>>>(x, W1, W2, Wa,
                                                     xbf, w1t, w2t, wat);
    }

    // Kernel 1: projections (WMMA + async global->LDS staging)
    hspa_proj_kernel<<<dim3(N_ / 64, B_), 256, 0, stream>>>(
        xbf, w1t, w2t, wat, b1, a1, b2, a2, ba, aa, e1, e2, vt);

    // Kernel 2: fused score + sparsemax + output (256KB LDS score stripe)
    const int smem_bytes = (16 * N_ + 64) * (int)sizeof(float);   // 256 KB + tau
    (void)hipFuncSetAttribute((const void*)hspa_attn_kernel,
                              hipFuncAttributeMaxDynamicSharedMemorySize,
                              smem_bytes);
    hspa_attn_kernel<<<dim3(N_ / 16, B_), 256, smem_bytes, stream>>>(
        e1, e2, vt, x, out);
}
